// Mamba2Layer_21388937134865
// MI455X (gfx1250) — matrix operand order repaired and checked by reading, not compiled, not run on hardware
//
#include <hip/hip_runtime.h>
#include <hip/hip_bf16.h>

// ---------------------------------------------------------------------------
// Mamba2 layer forward for MI455X (gfx1250, wave32, WMMA).
// GEMMs: v_wmma_f32_16x16x32_f16, f32 accumulate, 128x256 block tiles staged
// into double-buffered LDS with GLOBAL_LOAD_ASYNC_TO_LDS_B128 (ASYNCcnt) so
// the next k-tile streams in while the current one feeds the matrix pipes.
// Selective scan keeps the 64x128 per-head state in registers (32 f32/thread)
// and broadcasts per-step B/C/x through LDS.
// ---------------------------------------------------------------------------

typedef _Float16 half_t;
typedef half_t v8h  __attribute__((ext_vector_type(8)));
typedef half_t v16h __attribute__((ext_vector_type(16)));
typedef float  v8f  __attribute__((ext_vector_type(8)));

#define D_MODEL   1024
#define D_INNER   2048
#define D_STATE   128
#define NHEADS    32
#define HEADDIM   64
#define CONV_DIM  2304
#define D_IN_PROJ 4384
#define BATCH     2
#define SEQ       4096
#define NROWS     (BATCH * SEQ)     // 8192
#define EPS       1e-5f

// ------------------------------ helpers ------------------------------------

__device__ __forceinline__ float silu_f(float x) {
    return x / (1.f + __expf(-x));
}

// Async copy of 16 bytes global -> LDS, tracked by ASYNCcnt.
// LDS byte address = low 32 bits of the generic pointer (flat->LDS mapping
// uses addr[31:0]).
__device__ __forceinline__ void async_load_b128(void* lds_ptr, const void* gptr) {
    unsigned lds_addr = (unsigned)(size_t)lds_ptr;
    asm volatile("global_load_async_to_lds_b128 %0, %1, off"
                 :: "v"(lds_addr), "v"(gptr)
                 : "memory");
}

// 256-thread block sum; sbuf must hold >= 8 floats.
__device__ float block_sum_256(float v, float* sbuf) {
    const int t = threadIdx.x;
    #pragma unroll
    for (int off = 16; off; off >>= 1) v += __shfl_xor(v, off, 32);
    __syncthreads();                 // protect sbuf reuse across calls
    if ((t & 31) == 0) sbuf[t >> 5] = v;
    __syncthreads();
    if (t == 0) {
        float s = 0.f;
        #pragma unroll
        for (int i = 0; i < 8; ++i) s += sbuf[i];
        sbuf[0] = s;
    }
    __syncthreads();
    return sbuf[0];
}

// ------------------------------ casts --------------------------------------

__global__ void __launch_bounds__(256)
cast_f32_to_f16_kernel(const float* __restrict__ in, half_t* __restrict__ out, int n) {
    int i = blockIdx.x * 256 + threadIdx.x;
    if (i < n) out[i] = (half_t)in[i];
}

// ------------------------------ LayerNorm ----------------------------------

__global__ void __launch_bounds__(256)
layernorm_kernel(const float* __restrict__ x, const float* __restrict__ w,
                 const float* __restrict__ b, half_t* __restrict__ out) {
    __shared__ float sbuf[8];
    const int row = blockIdx.x;                 // b*SEQ + t
    const float* xr = x + (size_t)row * D_MODEL;
    float v[4];
    float s = 0.f;
    #pragma unroll
    for (int i = 0; i < 4; ++i) { v[i] = xr[threadIdx.x + i * 256]; s += v[i]; }
    s = block_sum_256(s, sbuf);
    const float mu = s * (1.f / D_MODEL);
    float sq = 0.f;
    #pragma unroll
    for (int i = 0; i < 4; ++i) { float d = v[i] - mu; sq += d * d; }
    sq = block_sum_256(sq, sbuf);
    const float inv = rsqrtf(sq * (1.f / D_MODEL) + EPS);
    half_t* orow = out + (size_t)row * D_MODEL;
    #pragma unroll
    for (int i = 0; i < 4; ++i) {
        int c = threadIdx.x + i * 256;
        orow[c] = (half_t)((v[i] - mu) * inv * w[c] + b[c]);
    }
}

// ------------------------------ WMMA GEMM ----------------------------------
// C[M,N] = A[M,K] * Bw[N,K]^T (+ resid), f16 inputs, f32 accumulate.
// Block tile 128M x 256N, 8 waves (2x4), wave tile 64x64 = 4x4 fragments.
// Double-buffered LDS fed by async global->LDS b128 loads (6 per thread per
// k-tile): s_wait_asynccnt 0x6 retires the previous tile while the next one
// stays in flight during the WMMAs.

#define TILE_K     32
#define LDS_STRIDE 40   // halves/row: 80B rows -> 16B aligned, spread banks

template<bool ADD_RESIDUAL>
__global__ void __launch_bounds__(256)
gemm_wmma_kernel(const half_t* __restrict__ A,    // [M,K]
                 const half_t* __restrict__ Bw,   // [N,K] row-major
                 const float*  __restrict__ resid,// [M,N] or nullptr
                 float* __restrict__ C,           // [M,N]
                 int M, int N, int K) {
    __shared__ __align__(16) half_t As[2][128 * LDS_STRIDE];  // 20 KB
    __shared__ __align__(16) half_t Bs[2][256 * LDS_STRIDE];  // 40 KB

    const int tid   = threadIdx.x;
    const int lane  = tid & 31;
    const int wave  = tid >> 5;         // 0..7
    const int wm    = wave & 1;         // 2 waves over M (64 each)
    const int wn    = wave >> 1;        // 4 waves over N (64 each)
    const int m0    = blockIdx.y * 128;
    const int n0    = blockIdx.x * 256;

    // staging maps
    const int arow  = tid >> 1;                  // 0..127
    const int ahalf = (tid & 1) << 4;            // 0 or 16 halves (32B)
    int brow = n0 + tid;                         // 0..255 rows of B tile
    if (brow >= N) brow = N - 1;                 // clamp (stores are guarded)

    // fragment-gather constants (CDNA5 wave32 16-bit layouts)
    const int l16   = lane & 15;
    const int hi16  = lane >> 4;        // 0 or 1
    const int a_kb  = hi16 * 8;         // A: K {0..7,16..23} / {8..15,24..31}
    const int b_kb  = hi16 * 8;         // B: K {0..7,16..23} / {8..15,24..31}

    v8f acc[4][4];
    #pragma unroll
    for (int fm = 0; fm < 4; ++fm)
        #pragma unroll
        for (int fn = 0; fn < 4; ++fn)
            #pragma unroll
            for (int r = 0; r < 8; ++r) acc[fm][fn][r] = 0.f;

    auto issue_tile = [&](int k0, int buf) {
        {   // A tile: 128 rows x 32 halves, 2 threads/row, 2 x b128 each
            const half_t* gp = A + (size_t)(m0 + arow) * K + k0 + ahalf;
            half_t* lp = &As[buf][arow * LDS_STRIDE + ahalf];
            async_load_b128(lp,     gp);
            async_load_b128(lp + 8, gp + 8);
        }
        {   // B tile: 256 rows x 32 halves, 1 thread/row, 4 x b128 each
            const half_t* gp = Bw + (size_t)brow * K + k0;
            half_t* lp = &Bs[buf][tid * LDS_STRIDE];
            async_load_b128(lp,      gp);
            async_load_b128(lp + 8,  gp + 8);
            async_load_b128(lp + 16, gp + 16);
            async_load_b128(lp + 24, gp + 24);
        }
    };

    const int T = K / TILE_K;
    issue_tile(0, 0);

    for (int it = 0; it < T; ++it) {
        const int cur = it & 1;
        // all waves finished reading buf[cur^1] (two tiles ago) -> safe target
        __syncthreads();
        if (it + 1 < T) {
            issue_tile((it + 1) * TILE_K, cur ^ 1);
            asm volatile("s_wait_asynccnt 0x6" ::: "memory");  // tile 'it' landed
        } else {
            asm volatile("s_wait_asynccnt 0x0" ::: "memory");
        }
        __syncthreads();   // every wave waited its own counter -> tile visible

        v16h af[4], bf[4];
        #pragma unroll
        for (int fm = 0; fm < 4; ++fm) {
            const int row  = wm * 64 + fm * 16 + l16;
            const int base = row * LDS_STRIDE + a_kb;
            v8h lo = *(const v8h*)&As[cur][base];        // K {0..7}/{8..15}
            v8h hi = *(const v8h*)&As[cur][base + 16];   // K {16..23}/{24..31}
            af[fm] = __builtin_shufflevector(lo, hi, 0,1,2,3,4,5,6,7,8,9,10,11,12,13,14,15);
        }
        #pragma unroll
        for (int fn = 0; fn < 4; ++fn) {
            const int row  = wn * 64 + fn * 16 + l16;
            const int base = row * LDS_STRIDE + b_kb;
            v8h lo = *(const v8h*)&Bs[cur][base];        // K {0..7}/{8..15}
            v8h hi = *(const v8h*)&Bs[cur][base + 16];   // K {16..23}/{24..31}
            bf[fn] = __builtin_shufflevector(lo, hi, 0,1,2,3,4,5,6,7,8,9,10,11,12,13,14,15);
        }
        #pragma unroll
        for (int fm = 0; fm < 4; ++fm)
            #pragma unroll
            for (int fn = 0; fn < 4; ++fn)
                acc[fm][fn] = __builtin_amdgcn_wmma_f32_16x16x32_f16(
                    false, af[fm], false, bf[fn],
                    (short)0, acc[fm][fn], false, false);
    }

    // ---- store: VGPR r -> M = r + hi16*8, N = l16
    #pragma unroll
    for (int fm = 0; fm < 4; ++fm) {
        #pragma unroll
        for (int fn = 0; fn < 4; ++fn) {
            const int col = n0 + wn * 64 + fn * 16 + l16;
            if (col < N) {
                const int mbase = m0 + wm * 64 + fm * 16 + hi16 * 8;
                #pragma unroll
                for (int r = 0; r < 8; ++r) {
                    const size_t idx = (size_t)(mbase + r) * N + col;
                    float v = acc[fm][fn][r];
                    if (ADD_RESIDUAL) v += resid[idx];
                    C[idx] = v;
                }
            }
        }
    }
}

// --------------------- conv1d(k=4, causal) + SiLU + dt ---------------------

__global__ void __launch_bounds__(256)
conv_silu_dt_kernel(const float* __restrict__ zxbcdt, const float* __restrict__ conv_w,
                    const float* __restrict__ conv_b, const float* __restrict__ dt_bias,
                    float* __restrict__ convOut, float* __restrict__ dtOut) {
    const int row = blockIdx.x;            // b*SEQ + t
    const int t   = row & (SEQ - 1);
    const size_t base = (size_t)row * D_IN_PROJ;
    for (int c = threadIdx.x; c < CONV_DIM; c += 256) {
        float acc = conv_b[c];
        #pragma unroll
        for (int j = 0; j < 4; ++j) {
            const int dt_ = j - 3;                    // tap at t-3+j
            if (t + dt_ >= 0)
                acc += zxbcdt[base + (ptrdiff_t)dt_ * D_IN_PROJ + D_INNER + c] * conv_w[c * 4 + j];
        }
        convOut[(size_t)row * CONV_DIM + c] = silu_f(acc);
    }
    if (threadIdx.x < NHEADS) {
        const int h = threadIdx.x;
        const float v = zxbcdt[base + (D_INNER + CONV_DIM) + h] + dt_bias[h];
        dtOut[(size_t)row * NHEADS + h] = (v > 20.f) ? v : log1pf(__expf(v));
    }
}

// ------------------------------ selective scan -----------------------------
// One workgroup per (batch, head). State h[64][128]: thread owns p=tid/4,
// n-quarter = tid%4 -> 32 f32 in registers. 4-lane shfl_xor reduces y.

__global__ void __launch_bounds__(256)
scan_kernel(const float* __restrict__ convOut, const float* __restrict__ dtBuf,
            const float* __restrict__ A_log, const float* __restrict__ Dp,
            float* __restrict__ yOut) {
    const int bh = blockIdx.x;
    const int b  = bh >> 5;
    const int h  = bh & 31;
    const float Ah = -__expf(A_log[h]);
    const float Dh = Dp[h];

    const int tid = threadIdx.x;
    const int p   = tid >> 2;
    const int q   = tid & 3;
    const int n0  = q * 32;

    __shared__ float sB[D_STATE], sC[D_STATE], sx[HEADDIM];
    __shared__ float sdt;

    float st[32];
    #pragma unroll
    for (int i = 0; i < 32; ++i) st[i] = 0.f;

    for (int t = 0; t < SEQ; ++t) {
        const size_t cb = ((size_t)b * SEQ + t) * CONV_DIM;
        if (tid < 128) sB[tid]        = convOut[cb + D_INNER + tid];
        else           sC[tid - 128]  = convOut[cb + D_INNER + D_STATE + (tid - 128)];
        if (tid < 64)  sx[tid]        = convOut[cb + h * HEADDIM + tid];
        if (tid == 0)  sdt            = dtBuf[((size_t)b * SEQ + t) * NHEADS + h];
        __syncthreads();

        const float dtk   = sdt;
        const float decay = __expf(dtk * Ah);
        const float coef  = dtk * sx[p];
        float yp = 0.f;
        #pragma unroll
        for (int i = 0; i < 32; ++i) {
            st[i] = st[i] * decay + coef * sB[n0 + i];
            yp   += st[i] * sC[n0 + i];
        }
        yp += __shfl_xor(yp, 1, 32);
        yp += __shfl_xor(yp, 2, 32);
        if (q == 0)
            yOut[((size_t)b * SEQ + t) * D_INNER + h * HEADDIM + p] = yp + Dh * sx[p];
        __syncthreads();
    }
}

// ------------------------- gated RMSNorm -> f16 ----------------------------

__global__ void __launch_bounds__(256)
gate_rmsnorm_kernel(const float* __restrict__ yIn, const float* __restrict__ zxbcdt,
                    const float* __restrict__ norm_w, half_t* __restrict__ yOut) {
    __shared__ float sbuf[8];
    const int row = blockIdx.x;
    const size_t yb = (size_t)row * D_INNER;
    const size_t zb = (size_t)row * D_IN_PROJ;
    float v[8];
    float sq = 0.f;
    #pragma unroll
    for (int i = 0; i < 8; ++i) {
        const int c = threadIdx.x + i * 256;
        const float g = yIn[yb + c] * silu_f(zxbcdt[zb + c]);   // z = first 2048 chans
        v[i] = g;
        sq += g * g;
    }
    sq = block_sum_256(sq, sbuf);
    const float inv = rsqrtf(sq * (1.f / D_INNER) + EPS);
    #pragma unroll
    for (int i = 0; i < 8; ++i) {
        const int c = threadIdx.x + i * 256;
        yOut[yb + c] = (half_t)(v[i] * inv * norm_w[c]);
    }
}

// ------------------------------ launcher -----------------------------------

static inline size_t align_up(size_t v, size_t a) { return (v + a - 1) & ~(a - 1); }

extern "C" void kernel_launch(void* const* d_in, const int* in_sizes, int n_in,
                              void* d_out, int out_size, void* d_ws, size_t ws_size,
                              hipStream_t stream) {
    const float* x          = (const float*)d_in[0];
    const float* ln_w       = (const float*)d_in[1];
    const float* ln_b       = (const float*)d_in[2];
    const float* in_proj_w  = (const float*)d_in[3];
    const float* conv_w     = (const float*)d_in[4];
    const float* conv_b     = (const float*)d_in[5];
    const float* dt_bias    = (const float*)d_in[6];
    const float* A_log      = (const float*)d_in[7];
    const float* Dp         = (const float*)d_in[8];
    const float* norm_w     = (const float*)d_in[9];
    const float* out_proj_w = (const float*)d_in[10];
    float* out = (float*)d_out;

    // ---- workspace carve-up (~335 MB)
    char* ws = (char*)d_ws;
    size_t off = 0;
    auto carve = [&](size_t bytes) { void* p = ws + off; off = align_up(off + bytes, 256); return p; };
    float*  zxbcdt = (float*) carve((size_t)NROWS * D_IN_PROJ * 4);   // 143.7 MB
    float*  convO  = (float*) carve((size_t)NROWS * CONV_DIM * 4);    //  75.5 MB
    float*  dtBuf  = (float*) carve((size_t)NROWS * NHEADS * 4);      //   1.0 MB
    float*  yBuf   = (float*) carve((size_t)NROWS * D_INNER * 4);     //  67.1 MB
    half_t* xn_h   = (half_t*)carve((size_t)NROWS * D_MODEL * 2);     //  16.8 MB
    half_t* yg_h   = (half_t*)carve((size_t)NROWS * D_INNER * 2);     //  33.6 MB
    half_t* wi_h   = (half_t*)carve((size_t)D_IN_PROJ * D_MODEL * 2); //   9.0 MB
    half_t* wo_h   = (half_t*)carve((size_t)D_MODEL * D_INNER * 2);   //   4.2 MB
    (void)ws_size; (void)n_in; (void)in_sizes; (void)out_size;

    // 1) weight casts f32 -> f16
    {
        int n1 = D_IN_PROJ * D_MODEL;
        cast_f32_to_f16_kernel<<<(n1 + 255) / 256, 256, 0, stream>>>(in_proj_w, wi_h, n1);
        int n2 = D_MODEL * D_INNER;
        cast_f32_to_f16_kernel<<<(n2 + 255) / 256, 256, 0, stream>>>(out_proj_w, wo_h, n2);
    }
    // 2) LayerNorm -> xn (f16)
    layernorm_kernel<<<NROWS, 256, 0, stream>>>(x, ln_w, ln_b, xn_h);
    // 3) in_proj GEMM: zxbcdt[8192,4384] = xn * Wi^T   (WMMA f16)
    gemm_wmma_kernel<false><<<dim3((D_IN_PROJ + 255) / 256, NROWS / 128), 256, 0, stream>>>(
        xn_h, wi_h, nullptr, zxbcdt, NROWS, D_IN_PROJ, D_MODEL);
    // 4) causal depthwise conv + SiLU + softplus(dt)
    conv_silu_dt_kernel<<<NROWS, 256, 0, stream>>>(zxbcdt, conv_w, conv_b, dt_bias, convO, dtBuf);
    // 5) selective scan (state in registers)
    scan_kernel<<<BATCH * NHEADS, 256, 0, stream>>>(convO, dtBuf, A_log, Dp, yBuf);
    // 6) gated RMSNorm -> yg (f16)
    gate_rmsnorm_kernel<<<NROWS, 256, 0, stream>>>(yBuf, zxbcdt, norm_w, yg_h);
    // 7) out_proj GEMM + residual: out[8192,1024] = x + yg * Wo^T   (WMMA f16)
    gemm_wmma_kernel<true><<<dim3(D_MODEL / 256, NROWS / 128), 256, 0, stream>>>(
        yg_h, wo_h, x, out, NROWS, D_MODEL, D_INNER);
}
